// ChunkedMultiHeadAttention_12421045420035
// MI455X (gfx1250) — compile-verified, matrix operand order checked by reading, not run on hardware
//
#include <hip/hip_runtime.h>
#include <hip/hip_bf16.h>
#include <math.h>

#define BB 2
#define TT 4096
#define DD 1024
#define HH 16
#define HD 64
#define MM (BB * TT)      // 8192 rows
#define NC (TT / 64)      // 64 query chunks

typedef __attribute__((ext_vector_type(16))) _Float16 v16h;
typedef __attribute__((ext_vector_type(8)))  _Float16 v8h;
typedef __attribute__((ext_vector_type(8)))  float    v8f;

// ---------------------------------------------------------------------------
// WMMA helpers (CDNA5 wave32, V_WMMA_F32_16X16X32_F16)
// ---------------------------------------------------------------------------
__device__ __forceinline__ v8f wmma_f16(v16h a, v16h b, v8f c) {
  return __builtin_amdgcn_wmma_f32_16x16x32_f16(false, a, false, b, (short)0, c,
                                                false, false);
}

__device__ __forceinline__ v8f zero_v8f() {
  v8f z;
#pragma unroll
  for (int i = 0; i < 8; ++i) z[i] = 0.0f;
  return z;
}

// ---------------------------------------------------------------------------
// CDNA5 async global->LDS DMA (ASYNCcnt-tracked, no VGPR round trip).
// lds = low 32 bits of the generic pointer (LDS byte offset per ISA aperture
// rules); gaddr = full 64-bit global address in a VGPR pair.
// ---------------------------------------------------------------------------
__device__ __forceinline__ void async_ld_b128(uint32_t lds, uint64_t gaddr) {
  asm volatile("global_load_async_to_lds_b128 %0, %1, off"
               :: "v"(lds), "v"(gaddr) : "memory");
}
__device__ __forceinline__ void wait_async0() {
  asm volatile("s_wait_asynccnt 0x0" ::: "memory");
}

// Load a 16x32 f16 A/B fragment from an LDS tile stored row-major [row][ld].
// CDNA5 layout: lanes 0-15 hold row (lane), K = 0..7 and 16..23;
//               lanes 16-31 hold the same rows, K = 8..15 and 24..31.
__device__ __forceinline__ v16h ld_frag(const _Float16* base, int ld,
                                        int rowBase, int kBase, int lane) {
  const _Float16* p =
      base + (size_t)(rowBase + (lane & 15)) * ld + kBase + ((lane >> 4) << 3);
  v16h f;
#pragma unroll
  for (int i = 0; i < 8; ++i) {
    f[i]     = p[i];
    f[i + 8] = p[i + 16];
  }
  return f;
}

// ---------------------------------------------------------------------------
// fp32 -> f16 cast
// ---------------------------------------------------------------------------
__global__ __launch_bounds__(256) void cast_f32_to_f16_kernel(
    const float* __restrict__ src, _Float16* __restrict__ dst, int n) {
  int i = blockIdx.x * 256 + threadIdx.x;
  if (i < n) dst[i] = (_Float16)src[i];
}

// ---------------------------------------------------------------------------
// GEMM core: C[128x128] tile of A[M x K] * W[N x K]^T (torch Linear layout).
// 256 threads = 8 waves in a 2(M) x 4(N) grid; each wave owns 64x32 = 4x2
// 16x16 fp32 accumulators. BK = 64 (two WMMA K-depths per staged tile),
// double-buffered LDS, async DMA staging overlapped with WMMA compute.
// ---------------------------------------------------------------------------
// Stage one 128x64-f16 tile: 256 threads x 32 halfs = 4 async b128 each.
__device__ __forceinline__ void stage_tile_async(const _Float16* __restrict__ g,
                                                 int ldg, int rowBase,
                                                 _Float16 (*lds)[72], int tid) {
  const int row = tid >> 1;
  const int seg = (tid & 1) * 32;
  const _Float16* gp = g + (size_t)(rowBase + row) * ldg + seg;
  _Float16* lp = &lds[row][seg];
#pragma unroll
  for (int i = 0; i < 4; ++i)
    async_ld_b128((uint32_t)(size_t)(lp + 8 * i),
                  (uint64_t)(size_t)(gp + 8 * i));
}

__device__ __forceinline__ void gemm_core(const _Float16* __restrict__ A,
                                          const _Float16* __restrict__ W,
                                          int Kd, int mBase, int nBase, int tid,
                                          _Float16 (*As)[128][72],
                                          _Float16 (*Ws)[128][72],
                                          v8f acc[4][2]) {
  const int lane = tid & 31, wid = tid >> 5;
  const int waveM = wid & 1, waveN = wid >> 1;

  stage_tile_async(A, Kd, mBase, As[0], tid);
  stage_tile_async(W, Kd, nBase, Ws[0], tid);
  wait_async0();
  __syncthreads();

#pragma unroll 1
  for (int k0 = 0; k0 < Kd; k0 += 64) {
    const int cur = (k0 >> 6) & 1, nxt = cur ^ 1;
    if (k0 + 64 < Kd) {  // overlap next tile's DMA with this tile's WMMAs
      stage_tile_async(A + k0 + 64, Kd, mBase, As[nxt], tid);
      stage_tile_async(W + k0 + 64, Kd, nBase, Ws[nxt], tid);
    }
#pragma unroll
    for (int kk = 0; kk < 64; kk += 32) {
      v16h av[4], bv[2];
#pragma unroll
      for (int mt = 0; mt < 4; ++mt)
        av[mt] = ld_frag(&As[cur][0][0], 72, waveM * 64 + mt * 16, kk, lane);
#pragma unroll
      for (int nt = 0; nt < 2; ++nt)
        bv[nt] = ld_frag(&Ws[cur][0][0], 72, waveN * 32 + nt * 16, kk, lane);
#pragma unroll
      for (int mt = 0; mt < 4; ++mt)
#pragma unroll
        for (int nt = 0; nt < 2; ++nt)
          acc[mt][nt] = wmma_f16(av[mt], bv[nt], acc[mt][nt]);
    }
    wait_async0();
    __syncthreads();
  }
}

// ---------------------------------------------------------------------------
// QKV projection: y = x @ W^T + b, scattered into attention layouts.
//   z==0 -> Q  [(b*H+h)*T + t]*64 + d
//   z==1 -> K  [(b*H+h)*T + t]*64 + d
//   z==2 -> V^T[(b*H+h)*64 + d]*T + t   (so P·V B-operand is row-contiguous)
// ---------------------------------------------------------------------------
__global__ __launch_bounds__(256) void qkv_gemm_kernel(
    const _Float16* __restrict__ Xh, const _Float16* __restrict__ Wqh,
    const _Float16* __restrict__ Wkh, const _Float16* __restrict__ Wvh,
    const float* __restrict__ bq, const float* __restrict__ bk,
    const float* __restrict__ bv, _Float16* __restrict__ Qo,
    _Float16* __restrict__ Ko, _Float16* __restrict__ VTo) {
  __shared__ _Float16 As[2][128][72];
  __shared__ _Float16 Ws[2][128][72];
  const int tid = threadIdx.x;
  const int mBase = blockIdx.y * 128, nBase = blockIdx.x * 128;
  const int z = blockIdx.z;
  const _Float16* W = (z == 0) ? Wqh : (z == 1) ? Wkh : Wvh;
  const float* bias = (z == 0) ? bq : (z == 1) ? bk : bv;

  v8f acc[4][2];
#pragma unroll
  for (int mt = 0; mt < 4; ++mt)
#pragma unroll
    for (int nt = 0; nt < 2; ++nt) acc[mt][nt] = zero_v8f();

  gemm_core(Xh, W, DD, mBase, nBase, tid, As, Ws, acc);

  const int lane = tid & 31, wid = tid >> 5;
  const int waveM = wid & 1, waveN = wid >> 1;
  const int hi = lane >> 4, ln = lane & 15;
#pragma unroll
  for (int mt = 0; mt < 4; ++mt)
#pragma unroll
    for (int nt = 0; nt < 2; ++nt) {
      const int col = nBase + waveN * 32 + nt * 16 + ln;  // output feature
      const int h = col >> 6, d = col & 63;
      const float bcol = bias[col];
#pragma unroll
      for (int r = 0; r < 8; ++r) {
        const int m = mBase + waveM * 64 + mt * 16 + r + 8 * hi;  // token row
        const int bb = m >> 12;          // m / T
        const int tt = m & (TT - 1);     // m % T
        const float v = acc[mt][nt][r] + bcol;
        const size_t bh = (size_t)bb * HH + h;
        if (z == 0)
          Qo[(bh * TT + tt) * HD + d] = (_Float16)v;
        else if (z == 1)
          Ko[(bh * TT + tt) * HD + d] = (_Float16)v;
        else
          VTo[(bh * HD + d) * TT + tt] = (_Float16)v;
      }
    }
}

// ---------------------------------------------------------------------------
// Chunked causal attention, flash-style online softmax.
// One workgroup per (query chunk of 64, b*H+h). 8 waves; S and O are 64x64 =
// 16 WMMA tiles, 2 per wave. O accumulators persist across key chunks.
// K/V^T chunks double-buffered; next chunk's async DMA overlaps S-WMMA,
// softmax and O-WMMA of the current chunk.
// ---------------------------------------------------------------------------
__device__ __forceinline__ void stage_chunk64_async(
    const _Float16* __restrict__ g, _Float16 (*lds)[72], int tid) {
  const int row = tid >> 2, seg = (tid & 3) * 16;  // 16 halfs per thread
  const _Float16* gp = g + (size_t)row * 0;        // placeholder (unused)
  (void)gp;
  // caller passes g already offset per-row via lambda-less helper below
}

__global__ __launch_bounds__(256) void attn_kernel(
    const _Float16* __restrict__ Q, const _Float16* __restrict__ Kb,
    const _Float16* __restrict__ VT, _Float16* __restrict__ Out) {
  const int qc = blockIdx.x;   // query chunk
  const int bh = blockIdx.y;   // b*H + h
  const int b = bh >> 4, h = bh & 15;

  __shared__ _Float16 Qs[64][72];
  __shared__ _Float16 Ks[2][64][72];
  __shared__ _Float16 Vts[2][64][72];   // [d][k]
  __shared__ _Float16 Ps[64][72];
  __shared__ float Sld[64][65];         // padded: conflict-free row scans
  __shared__ float mrow[64], lrow[64], rsrow[64];

  const int tid = threadIdx.x, lane = tid & 31, wid = tid >> 5;
  const int hi = lane >> 4, ln = lane & 15;
  const int row = tid >> 2, seg = (tid & 3) * 16;  // 16 halfs per thread

  // Async-stage the query chunk and the first K/V^T chunk.
  {
    const _Float16* qg = Q + ((size_t)bh * TT + qc * 64 + row) * HD + seg;
    async_ld_b128((uint32_t)(size_t)&Qs[row][seg], (uint64_t)(size_t)qg);
    async_ld_b128((uint32_t)(size_t)&Qs[row][seg + 8], (uint64_t)(size_t)(qg + 8));
    const _Float16* kg = Kb + ((size_t)bh * TT + row) * HD + seg;
    async_ld_b128((uint32_t)(size_t)&Ks[0][row][seg], (uint64_t)(size_t)kg);
    async_ld_b128((uint32_t)(size_t)&Ks[0][row][seg + 8], (uint64_t)(size_t)(kg + 8));
    const _Float16* vg = VT + ((size_t)bh * HD + row) * TT + seg;
    async_ld_b128((uint32_t)(size_t)&Vts[0][row][seg], (uint64_t)(size_t)vg);
    async_ld_b128((uint32_t)(size_t)&Vts[0][row][seg + 8], (uint64_t)(size_t)(vg + 8));
  }
  if (tid < 64) {
    mrow[tid] = -INFINITY;
    lrow[tid] = 0.0f;
  }

  // This wave's two 16x16 tiles of the 64x64 S/O grids.
  const int t0 = 2 * wid, t1 = 2 * wid + 1;
  const int tm0 = t0 >> 2, tn0 = t0 & 3;
  const int tm1 = t1 >> 2, tn1 = t1 & 3;
  v8f oacc0 = zero_v8f(), oacc1 = zero_v8f();

  wait_async0();
  __syncthreads();

  const float scale = 0.125f;  // 1/sqrt(64)

  for (int kc = 0; kc <= qc; ++kc) {
    const int cur = kc & 1, nxt = cur ^ 1;

    // Issue next chunk's DMA now; it overlaps all compute below.
    if (kc < qc) {
      const _Float16* kg = Kb + ((size_t)bh * TT + (kc + 1) * 64 + row) * HD + seg;
      async_ld_b128((uint32_t)(size_t)&Ks[nxt][row][seg], (uint64_t)(size_t)kg);
      async_ld_b128((uint32_t)(size_t)&Ks[nxt][row][seg + 8], (uint64_t)(size_t)(kg + 8));
      const _Float16* vg = VT + ((size_t)bh * HD + row) * TT + (kc + 1) * 64 + seg;
      async_ld_b128((uint32_t)(size_t)&Vts[nxt][row][seg], (uint64_t)(size_t)vg);
      async_ld_b128((uint32_t)(size_t)&Vts[nxt][row][seg + 8], (uint64_t)(size_t)(vg + 8));
    }

    // S = scale * Q . K^T  (contraction over d), masked, spilled to LDS.
#pragma unroll
    for (int idx = 0; idx < 2; ++idx) {
      const int tm = idx ? tm1 : tm0, tn = idx ? tn1 : tn0;
      v8f s = zero_v8f();
#pragma unroll
      for (int kk = 0; kk < 64; kk += 32) {
        v16h a = ld_frag(&Qs[0][0], 72, tm * 16, kk, lane);
        v16h bf = ld_frag(&Ks[cur][0][0], 72, tn * 16, kk, lane);
        s = wmma_f16(a, bf, s);
      }
      const int cl = tn * 16 + ln;          // local key col
      const int kpos = kc * 64 + cl;
#pragma unroll
      for (int r = 0; r < 8; ++r) {
        const int qrow = tm * 16 + r + 8 * hi;  // local query row
        float v = s[r] * scale;
        if (kpos > qc * 64 + qrow) v = -INFINITY;  // causal mask
        Sld[qrow][cl] = v;
      }
    }
    __syncthreads();

    // Online softmax update (one thread per query row).
    if (tid < 64) {
      const float mo = mrow[tid];
      float cm = -INFINITY;
#pragma unroll 4
      for (int j = 0; j < 64; ++j) cm = fmaxf(cm, Sld[tid][j]);
      const float nm = fmaxf(mo, cm);
      const float rs = __expf(mo - nm);  // 0 on the first chunk (mo = -inf)
      float sum = 0.0f;
#pragma unroll 4
      for (int j = 0; j < 64; ++j) {
        const float p = __expf(Sld[tid][j] - nm);
        sum += p;
        Ps[tid][j] = (_Float16)p;
      }
      lrow[tid] = lrow[tid] * rs + sum;
      mrow[tid] = nm;
      rsrow[tid] = rs;
    }
    __syncthreads();

    // O = O * rowscale + P . V   (contraction over k).
#pragma unroll
    for (int idx = 0; idx < 2; ++idx) {
      const int tm = idx ? tm1 : tm0, tn = idx ? tn1 : tn0;
      v8f o = idx ? oacc1 : oacc0;
#pragma unroll
      for (int r = 0; r < 8; ++r) o[r] *= rsrow[tm * 16 + r + 8 * hi];
#pragma unroll
      for (int kk = 0; kk < 64; kk += 32) {
        v16h a = ld_frag(&Ps[0][0], 72, tm * 16, kk, lane);
        v16h bf = ld_frag(&Vts[cur][0][0], 72, tn * 16, kk, lane);
        o = wmma_f16(a, bf, o);
      }
      if (idx) oacc1 = o; else oacc0 = o;
    }

    wait_async0();     // next chunk's DMA done
    __syncthreads();   // all waves past reads of cur / Sld / Ps
  }

  // Normalize by row sums and write attn output in (b, t, h*64+d) layout.
#pragma unroll
  for (int idx = 0; idx < 2; ++idx) {
    const int tm = idx ? tm1 : tm0, tn = idx ? tn1 : tn0;
    const v8f o = idx ? oacc1 : oacc0;
    const int d = tn * 16 + ln;
#pragma unroll
    for (int r = 0; r < 8; ++r) {
      const int qrow = tm * 16 + r + 8 * hi;
      const float v = o[r] / lrow[qrow];
      const int t = qc * 64 + qrow;
      Out[((size_t)b * TT + t) * DD + h * HD + d] = (_Float16)v;
    }
  }
}

// ---------------------------------------------------------------------------
// Output projection: out = attn @ Wo^T + bo, fp32 result.
// ---------------------------------------------------------------------------
__global__ __launch_bounds__(256) void oproj_gemm_kernel(
    const _Float16* __restrict__ Ah, const _Float16* __restrict__ Woh,
    const float* __restrict__ bo, float* __restrict__ out) {
  __shared__ _Float16 As[2][128][72];
  __shared__ _Float16 Ws[2][128][72];
  const int tid = threadIdx.x;
  const int mBase = blockIdx.y * 128, nBase = blockIdx.x * 128;

  v8f acc[4][2];
#pragma unroll
  for (int mt = 0; mt < 4; ++mt)
#pragma unroll
    for (int nt = 0; nt < 2; ++nt) acc[mt][nt] = zero_v8f();

  gemm_core(Ah, Woh, DD, mBase, nBase, tid, As, Ws, acc);

  const int lane = tid & 31, wid = tid >> 5;
  const int waveM = wid & 1, waveN = wid >> 1;
  const int hi = lane >> 4, ln = lane & 15;
#pragma unroll
  for (int mt = 0; mt < 4; ++mt)
#pragma unroll
    for (int nt = 0; nt < 2; ++nt) {
      const int col = nBase + waveN * 32 + nt * 16 + ln;
      const float bcol = bo[col];
#pragma unroll
      for (int r = 0; r < 8; ++r) {
        const int m = mBase + waveM * 64 + mt * 16 + r + 8 * hi;
        out[(size_t)m * DD + col] = acc[mt][nt][r] + bcol;
      }
    }
}

// ---------------------------------------------------------------------------
// Host launcher. Inputs: x, Wq, bq, Wk, bk, Wv, bv, Wo, bo (all fp32).
// ---------------------------------------------------------------------------
extern "C" void kernel_launch(void* const* d_in, const int* in_sizes, int n_in,
                              void* d_out, int out_size, void* d_ws,
                              size_t ws_size, hipStream_t stream) {
  const float* x  = (const float*)d_in[0];
  const float* Wq = (const float*)d_in[1];
  const float* bq = (const float*)d_in[2];
  const float* Wk = (const float*)d_in[3];
  const float* bk = (const float*)d_in[4];
  const float* Wv = (const float*)d_in[5];
  const float* bv = (const float*)d_in[6];
  const float* Wo = (const float*)d_in[7];
  const float* bo = (const float*)d_in[8];
  float* out = (float*)d_out;

  // Workspace layout (f16 elements).
  _Float16* ws = (_Float16*)d_ws;
  const size_t nX = (size_t)MM * DD;   // 8388608
  const size_t nW = (size_t)DD * DD;   // 1048576
  _Float16* Xh   = ws;
  _Float16* Wqh  = Xh + nX;
  _Float16* Wkh  = Wqh + nW;
  _Float16* Wvh  = Wkh + nW;
  _Float16* Woh  = Wvh + nW;
  _Float16* Qh   = Woh + nW;
  _Float16* Kh   = Qh + nX;
  _Float16* VTh  = Kh + nX;
  _Float16* AttH = VTh + nX;

  // Casts.
  cast_f32_to_f16_kernel<<<(int)((nX + 255) / 256), 256, 0, stream>>>(x, Xh, (int)nX);
  cast_f32_to_f16_kernel<<<(int)((nW + 255) / 256), 256, 0, stream>>>(Wq, Wqh, (int)nW);
  cast_f32_to_f16_kernel<<<(int)((nW + 255) / 256), 256, 0, stream>>>(Wk, Wkh, (int)nW);
  cast_f32_to_f16_kernel<<<(int)((nW + 255) / 256), 256, 0, stream>>>(Wv, Wvh, (int)nW);
  cast_f32_to_f16_kernel<<<(int)((nW + 255) / 256), 256, 0, stream>>>(Wo, Woh, (int)nW);

  // QKV projections (grid.z selects Q/K/V).
  dim3 gq(DD / 128, MM / 128, 3);
  qkv_gemm_kernel<<<gq, 256, 0, stream>>>(Xh, Wqh, Wkh, Wvh, bq, bk, bv, Qh, Kh, VTh);

  // Chunked causal attention.
  dim3 ga(NC, BB * HH, 1);
  attn_kernel<<<ga, 256, 0, stream>>>(Qh, Kh, VTh, AttH);

  // Output projection.
  dim3 go(DD / 128, MM / 128, 1);
  oproj_gemm_kernel<<<go, 256, 0, stream>>>(AttH, Woh, bo, out);
}
